// MultiHeadAttention_65893388255704
// MI455X (gfx1250) — compile-verified, matrix operand order checked by reading, not run on hardware
//
#include <hip/hip_runtime.h>
#include <hip/hip_bf16.h>
#include <stdint.h>

typedef __attribute__((ext_vector_type(16))) __bf16 v16bf;
typedef __attribute__((ext_vector_type(8)))  __bf16 v8bf;
typedef __attribute__((ext_vector_type(8)))  float  v8f;

#define D_MODEL 1024
#define NHEADS  16
#define HDIM    64
#define SEQ     2048
#define BATCH   2
#define NROWS   (BATCH * SEQ)   // 4096
#define E3      (3 * D_MODEL)   // 3072

__device__ __forceinline__ unsigned short f32_bf16(float f) {
  unsigned int u = __float_as_uint(f);
  u += 0x7FFFu + ((u >> 16) & 1u);   // round-to-nearest-even
  return (unsigned short)(u >> 16);
}

// Async global->LDS copy of 16B per lane (CDNA5 GLOBAL_LOAD_ASYNC_TO_LDS_B128,
// tracked by ASYNCcnt). ldsaddr = per-lane LDS byte address (low 32 bits of the
// flat shared pointer), gaddr = per-lane global address.
__device__ __forceinline__ void async_cp16(unsigned ldsaddr,
                                           unsigned long long gaddr) {
  asm volatile("global_load_async_to_lds_b128 %0, %1, off" ::"v"(ldsaddr),
               "v"(gaddr)
               : "memory");
}
__device__ __forceinline__ void async_wait0() {
  asm volatile("s_wait_asynccnt 0x0" ::: "memory");
}
__device__ __forceinline__ unsigned lds_off(const void* p) {
  return (unsigned)(unsigned long long)p;  // flat LDS addr: low 32 bits = offset
}

// 16-lane butterfly reductions. Prefer v_permlane16_b32 (VALU lane crossbar,
// no LDS traffic) over ds_bpermute-based shuffles.
#if __has_builtin(__builtin_amdgcn_permlane16)
__device__ __forceinline__ float perm16(float v, unsigned lo, unsigned hi) {
  return __uint_as_float(__builtin_amdgcn_permlane16(
      __float_as_uint(v), __float_as_uint(v), lo, hi, false, false));
}
__device__ __forceinline__ float red_max16(float v) {
  v = fmaxf(v, perm16(v, 0x67452301u, 0xEFCDAB89u));  // xor 1
  v = fmaxf(v, perm16(v, 0x54761032u, 0xDCFE98BAu));  // xor 2
  v = fmaxf(v, perm16(v, 0x32107654u, 0xBA98FEDCu));  // xor 4
  v = fmaxf(v, perm16(v, 0xFEDCBA98u, 0x76543210u));  // xor 8
  return v;
}
__device__ __forceinline__ float red_sum16(float v) {
  v += perm16(v, 0x67452301u, 0xEFCDAB89u);
  v += perm16(v, 0x54761032u, 0xDCFE98BAu);
  v += perm16(v, 0x32107654u, 0xBA98FEDCu);
  v += perm16(v, 0xFEDCBA98u, 0x76543210u);
  return v;
}
#else
__device__ __forceinline__ float red_max16(float v) {
  for (int off = 1; off < 16; off <<= 1) v = fmaxf(v, __shfl_xor(v, off, 32));
  return v;
}
__device__ __forceinline__ float red_sum16(float v) {
  for (int off = 1; off < 16; off <<= 1) v += __shfl_xor(v, off, 32);
  return v;
}
#endif

// Build a 16x32 bf16 A-operand (or B^T-operand) fragment from LDS.
// Lane layout per CDNA5 ISA 7.12.2 (16-bit A 16x32):
//   lanes 0-15 : row = lane,    K = [0..7]  and [16..23]
//   lanes 16-31: row = lane-16, K = [8..15] and [24..31]
// Caller passes row already including (lane & 15).
__device__ __forceinline__ v16bf frag_ld(const unsigned short* lds, int row,
                                         int ld, int kcol, int lane) {
  int kh = (lane >> 4) << 3;   // 0 or 8
  const v8bf* p0 = reinterpret_cast<const v8bf*>(lds + row * ld + kcol + kh);
  const v8bf* p1 = reinterpret_cast<const v8bf*>(lds + row * ld + kcol + 16 + kh);
  v8bf lo = *p0;
  v8bf hi = *p1;
  return __builtin_shufflevector(lo, hi, 0, 1, 2, 3, 4, 5, 6, 7,
                                         8, 9, 10, 11, 12, 13, 14, 15);
}

__global__ void cvt_f32_bf16(const float* __restrict__ in,
                             unsigned short* __restrict__ out, int n) {
  int i = blockIdx.x * blockDim.x + threadIdx.x;
  int stride = gridDim.x * blockDim.x;
  for (; i < n; i += stride) out[i] = f32_bf16(in[i]);
}

// C[M,N] = A[M,K] * W[N,K]^T + bias ; A,W bf16, accumulate f32.
// Block: 256 threads (8 waves). Tile 128x128, BK=32, double-buffered LDS fed
// by async global->LDS DMA: next tile's async copies are issued before the
// current tile's WMMAs, drained with s_wait_asynccnt + one barrier per K-step.
template <bool OUT_BF16>
__global__ __launch_bounds__(256) void gemm_bf16(
    const unsigned short* __restrict__ A, const unsigned short* __restrict__ W,
    const float* __restrict__ bias, void* __restrict__ outp, int Mrows, int N,
    int K) {
  const int LD = 40;  // 80B row stride: 16 rows -> 16 distinct banks, 16B aligned
  __shared__ __align__(16) unsigned short As[2][128 * LD];
  __shared__ __align__(16) unsigned short Ws[2][128 * LD];

  const int t = threadIdx.x;
  const int lane = t & 31;
  const int wave = t >> 5;   // 0..7
  const int wm = wave >> 2;  // 0..1
  const int wn = wave & 3;   // 0..3
  const int m0 = blockIdx.y * 128;
  const int n0 = blockIdx.x * 128;
  const int steps = K / 32;

  v8f acc[4][2];
#pragma unroll
  for (int mi = 0; mi < 4; mi++)
#pragma unroll
    for (int ni = 0; ni < 2; ni++) acc[mi][ni] = {};

  auto issue_tile = [&](int buf, int k0) {
#pragma unroll
    for (int i = 0; i < 2; i++) {
      int idx = t + i * 256;
      int row = idx >> 2, seg = idx & 3;
      async_cp16(lds_off(&As[buf][row * LD + seg * 8]),
                 (unsigned long long)&A[(size_t)(m0 + row) * K + k0 + seg * 8]);
      async_cp16(lds_off(&Ws[buf][row * LD + seg * 8]),
                 (unsigned long long)&W[(size_t)(n0 + row) * K + k0 + seg * 8]);
    }
  };

  issue_tile(0, 0);
  async_wait0();
  __syncthreads();

  for (int ks = 0; ks < steps; ks++) {
    const int cur = ks & 1, nxt = cur ^ 1;
    const bool more = (ks + 1) < steps;
    if (more) {  // DMA next tile into the alternate buffer during the WMMAs
      issue_tile(nxt, (ks + 1) * 32);
      if (ks + 2 < steps) {  // warm L2 one more tile ahead
        __builtin_prefetch(&A[(size_t)(m0 + (t >> 1)) * K + (ks + 2) * 32], 0, 0);
        __builtin_prefetch(&W[(size_t)(n0 + (t >> 1)) * K + (ks + 2) * 32], 0, 0);
      }
    }

    v16bf af[4], bfr[2];
#pragma unroll
    for (int mi = 0; mi < 4; mi++)
      af[mi] = frag_ld(As[cur], wm * 64 + mi * 16 + (lane & 15), LD, 0, lane);
#pragma unroll
    for (int ni = 0; ni < 2; ni++)
      bfr[ni] = frag_ld(Ws[cur], wn * 32 + ni * 16 + (lane & 15), LD, 0, lane);
#pragma unroll
    for (int mi = 0; mi < 4; mi++)
#pragma unroll
      for (int ni = 0; ni < 2; ni++)
        acc[mi][ni] = __builtin_amdgcn_wmma_f32_16x16x32_bf16(
            false, af[mi], false, bfr[ni], (short)0, acc[mi][ni], false, false);

    if (more) async_wait0();
    __syncthreads();
  }

  // C layout: VGPR r, lane l -> row = r + 8*(l>>4), col = l&15
  const int rbase = (lane >> 4) << 3;
  const int col0 = n0 + wn * 32 + (lane & 15);
#pragma unroll
  for (int ni = 0; ni < 2; ni++) {
    int c = col0 + ni * 16;
    float bv = bias[c];
#pragma unroll
    for (int mi = 0; mi < 4; mi++) {
#pragma unroll
      for (int r = 0; r < 8; r++) {
        size_t row = (size_t)(m0 + wm * 64 + mi * 16 + rbase + r);
        float v = acc[mi][ni][r] + bv;
        if (OUT_BF16)
          ((unsigned short*)outp)[row * N + c] = f32_bf16(v);
        else
          ((float*)outp)[row * N + c] = v;
      }
    }
  }
}

// Flash attention: one block per (b, h, 64-row q-tile). 128 threads = 4 waves,
// each wave owns 16 q rows. Streams keys in 64-wide chunks: K via async
// global->LDS DMA into double buffers, V register-staged (transposed on the
// way into LDS), one barrier per chunk.
__global__ __launch_bounds__(128) void attn_kernel(
    const unsigned short* __restrict__ QKV,  // [NROWS, 3072] bf16
    unsigned short* __restrict__ O) {        // [NROWS, 1024] bf16
  const int LD = 72;  // 144B stride: conflict-free 16-row access, 16B aligned
  __shared__ __align__(16) unsigned short Qs[64 * LD];
  __shared__ __align__(16) unsigned short Ks[2][64 * LD];
  __shared__ __align__(16) unsigned short Vts[2][64 * LD];  // [hd][key]
  __shared__ __align__(16) unsigned short Ps[64 * LD];

  const int t = threadIdx.x;
  const int lane = t & 31;
  const int wave = t >> 5;  // 0..3
  const int nq = SEQ / 64;
  const int bh = blockIdx.x / nq;
  const int qt = blockIdx.x % nq;
  const int b = bh / NHEADS, h = bh % NHEADS;
  const size_t rowbase = (size_t)b * SEQ;
  const int qcol = h * 192, kcol = h * 192 + 64, vcol = h * 192 + 128;
  const int NC = SEQ / 64;

  auto issue_k = [&](int buf, int kc) {
#pragma unroll
    for (int i = 0; i < 4; i++) {
      int idx = t + i * 128;
      int row = idx >> 3, seg = idx & 7;
      async_cp16(
          lds_off(&Ks[buf][row * LD + seg * 8]),
          (unsigned long long)&QKV[(rowbase + (size_t)kc * 64 + row) * E3 +
                                   kcol + seg * 8]);
    }
  };

  uint4 rv[4];
  auto load_v = [&](int kc) {
#pragma unroll
    for (int i = 0; i < 4; i++) {
      int idx = t + i * 128;
      int row = idx >> 3, seg = idx & 7;
      rv[i] = *reinterpret_cast<const uint4*>(
          &QKV[(rowbase + (size_t)kc * 64 + row) * E3 + vcol + seg * 8]);
    }
  };
  auto store_v = [&](int buf) {
#pragma unroll
    for (int i = 0; i < 4; i++) {
      int idx = t + i * 128;
      int row = idx >> 3, seg = idx & 7;
      const unsigned short* vs = reinterpret_cast<const unsigned short*>(&rv[i]);
#pragma unroll
      for (int j = 0; j < 8; j++) Vts[buf][(seg * 8 + j) * LD + row] = vs[j];
    }
  };

  // Prologue: Q tile (async), K chunk 0 (async), V chunk 0 (registers)
#pragma unroll
  for (int i = 0; i < 4; i++) {
    int idx = t + i * 128;
    int row = idx >> 3, seg = idx & 7;
    async_cp16(lds_off(&Qs[row * LD + seg * 8]),
               (unsigned long long)&QKV[(rowbase + (size_t)qt * 64 + row) * E3 +
                                        qcol + seg * 8]);
  }
  issue_k(0, 0);
  load_v(0);
  store_v(0);
  async_wait0();
  __syncthreads();

  float run_m[8], run_l[8];
  v8f o[4];
#pragma unroll
  for (int r = 0; r < 8; r++) {
    run_m[r] = -1e30f;
    run_l[r] = 0.f;
  }
#pragma unroll
  for (int d = 0; d < 4; d++) o[d] = {};

  const float scale = 0.125f;  // 1/sqrt(64)
  const float LOG2E = 1.44269504f;
  const int rbase = (lane >> 4) << 3;
  const int cidx = lane & 15;

  for (int kc = 0; kc < NC; kc++) {
    const int cur = kc & 1, nxt = cur ^ 1;
    const bool more = (kc + 1) < NC;
    if (more) {  // next chunk: K via DMA, V into registers
      issue_k(nxt, kc + 1);
      load_v(kc + 1);
      if (kc + 2 < NC) {
        size_t pr = rowbase + (size_t)(kc + 2) * 64 + (t >> 1);
        __builtin_prefetch(&QKV[pr * E3 + kcol + (t & 1) * 32], 0, 0);
        __builtin_prefetch(&QKV[pr * E3 + vcol + (t & 1) * 32], 0, 0);
      }
    }

    // S = Q * K^T  (16 q-rows x 64 keys per wave)
    v8f sc[4];
#pragma unroll
    for (int ni = 0; ni < 4; ni++) sc[ni] = {};
#pragma unroll
    for (int ks = 0; ks < 2; ks++) {
      v16bf aq = frag_ld(Qs, wave * 16 + cidx, LD, ks * 32, lane);
#pragma unroll
      for (int ni = 0; ni < 4; ni++) {
        v16bf bk = frag_ld(Ks[cur], ni * 16 + cidx, LD, ks * 32, lane);
        sc[ni] = __builtin_amdgcn_wmma_f32_16x16x32_bf16(
            false, aq, false, bk, (short)0, sc[ni], false, false);
      }
    }

    // Online softmax (row stats replicated across each 16-lane group)
#pragma unroll
    for (int r = 0; r < 8; r++) {
      float mx = -1e30f;
#pragma unroll
      for (int ni = 0; ni < 4; ni++) {
        sc[ni][r] *= scale;
        mx = fmaxf(mx, sc[ni][r]);
      }
      mx = red_max16(mx);
      float m_new = fmaxf(run_m[r], mx);
      float corr = exp2f((run_m[r] - m_new) * LOG2E);
      float rs = 0.f;
#pragma unroll
      for (int ni = 0; ni < 4; ni++) {
        float p = exp2f((sc[ni][r] - m_new) * LOG2E);
        sc[ni][r] = p;
        rs += p;
      }
      rs = red_sum16(rs);
      run_l[r] = run_l[r] * corr + rs;
      run_m[r] = m_new;
#pragma unroll
      for (int d = 0; d < 4; d++) o[d][r] *= corr;
    }

    // P: C-layout -> LDS -> A-layout (per-wave private region)
#pragma unroll
    for (int r = 0; r < 8; r++)
#pragma unroll
      for (int ni = 0; ni < 4; ni++)
        Ps[(wave * 16 + rbase + r) * LD + ni * 16 + cidx] = f32_bf16(sc[ni][r]);
    asm volatile("s_wait_dscnt 0" ::: "memory");

    // O += P * V  (B operand = V^T rows, contiguous in Vts)
#pragma unroll
    for (int ks = 0; ks < 2; ks++) {
      v16bf ap = frag_ld(Ps, wave * 16 + cidx, LD, ks * 32, lane);
#pragma unroll
      for (int d = 0; d < 4; d++) {
        v16bf bv = frag_ld(Vts[cur], d * 16 + cidx, LD, ks * 32, lane);
        o[d] = __builtin_amdgcn_wmma_f32_16x16x32_bf16(
            false, ap, false, bv, (short)0, o[d], false, false);
      }
    }

    if (more) {
      store_v(nxt);    // drain V registers into the alternate buffer
      async_wait0();   // K DMA for the alternate buffer complete
    }
    __syncthreads();
  }

  // Normalize + store bf16 attention output in [B,S,H*hd] layout
#pragma unroll
  for (int r = 0; r < 8; r++) {
    float inv = 1.f / run_l[r];
    size_t row = rowbase + qt * 64 + wave * 16 + rbase + r;
#pragma unroll
    for (int d = 0; d < 4; d++)
      O[row * D_MODEL + h * HDIM + d * 16 + cidx] = f32_bf16(o[d][r] * inv);
  }
}

extern "C" void kernel_launch(void* const* d_in, const int* in_sizes, int n_in,
                              void* d_out, int out_size, void* d_ws,
                              size_t ws_size, hipStream_t stream) {
  const float* x     = (const float*)d_in[0];
  const float* qkv_w = (const float*)d_in[1];
  const float* qkv_b = (const float*)d_in[2];
  const float* out_w = (const float*)d_in[3];
  const float* out_b = (const float*)d_in[4];

  char* ws = (char*)d_ws;
  unsigned short* Xh    = (unsigned short*)(ws);              // 8 MB
  unsigned short* Wqkvh = (unsigned short*)(ws + 8388608);    // 6 MB
  unsigned short* Wouth = (unsigned short*)(ws + 14680064);   // 2 MB
  unsigned short* QKVh  = (unsigned short*)(ws + 16777216);   // 24 MB
  unsigned short* Oh    = (unsigned short*)(ws + 41943040);   // 8 MB

  cvt_f32_bf16<<<4096, 256, 0, stream>>>(x, Xh, NROWS * D_MODEL);
  cvt_f32_bf16<<<3072, 256, 0, stream>>>(qkv_w, Wqkvh, E3 * D_MODEL);
  cvt_f32_bf16<<<1024, 256, 0, stream>>>(out_w, Wouth, D_MODEL * D_MODEL);

  gemm_bf16<true><<<dim3(E3 / 128, NROWS / 128), 256, 0, stream>>>(
      Xh, Wqkvh, qkv_b, (void*)QKVh, NROWS, E3, D_MODEL);

  attn_kernel<<<BATCH * NHEADS * (SEQ / 64), 128, 0, stream>>>(QKVh, Oh);

  gemm_bf16<false><<<dim3(D_MODEL / 128, NROWS / 128), 256, 0, stream>>>(
      Oh, Wouth, out_b, d_out, NROWS, D_MODEL, D_MODEL);
}
